// LongformerICDClassifier_63806034149418
// MI455X (gfx1250) — matrix operand order repaired and checked by reading, not compiled, not run on hardware
//
#include <hip/hip_runtime.h>

// ---------------- model constants ----------------
#define LAYERS 2
#define HID    768
#define HEADS  12
#define DH     64
#define WW     256          // one-sided window == chunk size
#define SEQ    4096
#define BB     2
#define NC     (SEQ / WW)   // 16 chunks
#define FFN    3072
#define NLAB   8921
#define NLABP  8960         // NLAB padded to multiple of 128 for aligned B tiles
#define TOK    (BB * SEQ)   // 8192 tokens
#define NEGBIG (-1e9f)

typedef unsigned short u16;
typedef __attribute__((ext_vector_type(8)))  u16    v8u;
typedef __attribute__((ext_vector_type(16))) __bf16 v16bf;
typedef __attribute__((ext_vector_type(8)))  float  v8f;
typedef __attribute__((ext_vector_type(4)))  int    v4i;

typedef __attribute__((address_space(1))) v4i*  as1_v4ip;  // global int4*
typedef __attribute__((address_space(3))) void* as3_vp;    // LDS void*
typedef __attribute__((address_space(3))) v4i*  as3_v4ip;  // LDS int4*

// CDNA5 async global->LDS copy (ASYNCcnt) with portable fallback
#if defined(__has_builtin)
#if __has_builtin(__builtin_amdgcn_global_load_async_to_lds_b128) && \
    __has_builtin(__builtin_amdgcn_s_wait_asynccnt)
#define LF_ASYNC 1
#endif
#endif

// ---------------- small helpers ----------------
static __device__ __forceinline__ u16 f2bf(float f) {
  unsigned int u = __builtin_bit_cast(unsigned int, f);
  u += 0x7fffu + ((u >> 16) & 1u);          // round-to-nearest-even
  return (u16)(u >> 16);
}
static __device__ __forceinline__ float bf2f(u16 h) {
  unsigned int u = ((unsigned int)h) << 16;
  return __builtin_bit_cast(float, u);
}
union FragCvt { struct { v8u lo, hi; } u; v16bf f; };
static __device__ __forceinline__ v16bf mkfrag(v8u lo, v8u hi) {
  FragCvt c; c.u.lo = lo; c.u.hi = hi; return c.f;
}
static __device__ __forceinline__ v8f wmma_bf16(v16bf a, v16bf b, v8f c) {
  return __builtin_amdgcn_wmma_f32_16x16x32_bf16(false, a, false, b, (short)0, c,
                                                 false, false);
}
// 16-byte global -> LDS copy: async data mover if available, else load+store
static __device__ __forceinline__ void cp16(u16* dstLds, const u16* srcGlb) {
#ifdef LF_ASYNC
  __builtin_amdgcn_global_load_async_to_lds_b128(
      (as1_v4ip)(unsigned long long)srcGlb, (as3_v4ip)(as3_vp)dstLds, 0, 0);
#else
  *(v8u*)dstLds = *(const v8u*)srcGlb;
#endif
}
static __device__ __forceinline__ void lf_wait_async() {
#ifdef LF_ASYNC
  __builtin_amdgcn_s_wait_asynccnt(0);
#endif
}

// ---------------- fp32 -> bf16 weight conversion ----------------
__global__ __launch_bounds__(256) void lf_cvt_bf16(const float* __restrict__ x,
                                                   u16* __restrict__ y, long n) {
  long i = (long)blockIdx.x * 256 + threadIdx.x;
  if (i < n) y[i] = f2bf(x[i]);
}
// conversion with zero-padded leading dimension (for the 8921-label head)
__global__ __launch_bounds__(256) void lf_cvt_pad_bf16(const float* __restrict__ x,
                                                       u16* __restrict__ y,
                                                       int rows, int cols, int ldy) {
  long i = (long)blockIdx.x * 256 + threadIdx.x;
  if (i < (long)rows * ldy) {
    int r = (int)(i / ldy), c = (int)(i % ldy);
    y[i] = (c < cols) ? f2bf(x[(size_t)r * cols + c]) : (u16)0;
  }
}

// ---------------- generic bf16 WMMA GEMM: C = A[M,K] @ B[K,N](ld=ldB) + bias --
// mode 0: store fp32 to Cf; mode 1: store bf16 to Cb; mode 2: exact GELU -> Cb
__global__ __launch_bounds__(256) void lf_gemm_bf16(
    const u16* __restrict__ A, const u16* __restrict__ Bw,
    const float* __restrict__ bias, float* __restrict__ Cf, u16* __restrict__ Cb,
    int M, int N, int K, int ldB, int mode) {
  __shared__ u16 As[64][40];    // 64x32 tile, pad to 40 (16B-aligned frag rows)
  __shared__ u16 Bs[32][136];   // 32x128 tile, pad to 136
  const int tid = threadIdx.x;
  const int wv = tid >> 5, lane = tid & 31;
  const int wr = wv >> 2, wc = wv & 3;   // 2x4 wave grid, 32x32 per wave
  const int m0 = blockIdx.y * 64;
  const int n0 = blockIdx.x * 128;

  // fixed per-thread staging coordinates: A 1x16B, B 2x16B per K-step
  const int ar = tid >> 2, ac = (tid & 3) << 3;       // A: 64 rows x 4 packets
  int agr = m0 + ar; if (agr >= M) agr = M - 1;
  const u16* aptr = A + (size_t)agr * K + ac;
  const int br = tid >> 3, bc = (tid & 7) << 4;       // B: 32 rows x 8 packets
  const u16* bptr = Bw + (size_t)br * ldB + n0 + bc;

  v8f acc[2][2];
#pragma unroll
  for (int t = 0; t < 2; ++t)
#pragma unroll
    for (int u = 0; u < 2; ++u) acc[t][u] = (v8f){};

  for (int k0 = 0; k0 < K; k0 += 32) {
    __syncthreads();
    cp16(&As[ar][ac], aptr + k0);
    {
      const u16* bsrc = bptr + (size_t)k0 * ldB;
      cp16(&Bs[br][bc], bsrc);
      cp16(&Bs[br][bc + 8], bsrc + 8);
    }
    if (k0 + 32 < K) {                    // gfx1250 global_prefetch_b8
      __builtin_prefetch(aptr + k0 + 32, 0, 1);
      __builtin_prefetch(bptr + (size_t)(k0 + 32) * ldB, 0, 1);
    }
    lf_wait_async();
    __syncthreads();

    v16bf af[2], bfr[2];
#pragma unroll
    for (int t = 0; t < 2; ++t) {         // A fragments (16x32)
      int row = wr * 32 + t * 16 + (lane & 15);
      int ko = (lane >> 4) << 3;
      af[t] = mkfrag(*(const v8u*)&As[row][ko], *(const v8u*)&As[row][ko + 16]);
    }
#pragma unroll
    for (int u = 0; u < 2; ++u) {         // B fragments (32x16), lane == N column
      int cc = wc * 32 + u * 16 + (lane & 15);
#pragma unroll
      for (int e = 0; e < 16; ++e)
        bfr[u][e] = __builtin_bit_cast(__bf16, Bs[((lane >> 4) << 3) + (e & 7) + ((e >> 3) << 4)][cc]);
    }
#pragma unroll
    for (int t = 0; t < 2; ++t)
#pragma unroll
      for (int u = 0; u < 2; ++u)
        acc[t][u] = wmma_bf16(af[t], bfr[u], acc[t][u]);
  }

#pragma unroll
  for (int t = 0; t < 2; ++t)
#pragma unroll
    for (int u = 0; u < 2; ++u)
#pragma unroll
      for (int r = 0; r < 8; ++r) {
        int grow = m0 + wr * 32 + t * 16 + r + ((lane >> 4) << 3);
        int gcol = n0 + wc * 32 + u * 16 + (lane & 15);
        if (grow < M && gcol < N) {
          float vv = acc[t][u][r];
          if (bias) vv += bias[gcol];
          if (mode == 2) vv = 0.5f * vv * (1.0f + erff(vv * 0.70710678118654752f));
          if (mode == 0) Cf[(size_t)grow * N + gcol] = vv;
          else           Cb[(size_t)grow * N + gcol] = f2bf(vv);
        }
      }
}

// ---------------- embeddings: yo = word_emb[ids] + pos_emb ----------------
__global__ __launch_bounds__(256) void lf_embed_add(
    const int* __restrict__ ids, const float* __restrict__ wemb,
    const float* __restrict__ pemb, float* __restrict__ yo) {
  const int tok = blockIdx.x, tid = threadIdx.x;
  const int id = ids[tok];
  const int s = tok % SEQ;
  for (int i = tid; i < HID; i += 256)
    yo[(size_t)tok * HID + i] = wemb[(size_t)id * HID + i] + pemb[(size_t)s * HID + i];
}

// ---------------- residual + LayerNorm -> fp32 Xf and bf16 Xh ----------------
__global__ __launch_bounds__(256) void lf_resid_ln(
    const float* __restrict__ X, const float* __restrict__ Y,
    const float* __restrict__ sc, const float* __restrict__ bs,
    float* __restrict__ Xf, u16* __restrict__ Xh) {
  __shared__ float red[256];
  const int row = blockIdx.x, tid = threadIdx.x;
  float vals[3], sum = 0.f;
#pragma unroll
  for (int j = 0; j < 3; ++j) {
    int i = tid + j * 256;
    float v = Y[(size_t)row * HID + i];
    if (X) v += X[(size_t)row * HID + i];
    vals[j] = v; sum += v;
  }
  red[tid] = sum; __syncthreads();
  for (int off = 128; off; off >>= 1) { if (tid < off) red[tid] += red[tid + off]; __syncthreads(); }
  const float mean = red[0] * (1.0f / HID);
  __syncthreads();
  float ss = 0.f;
#pragma unroll
  for (int j = 0; j < 3; ++j) { float d = vals[j] - mean; ss += d * d; }
  red[tid] = ss; __syncthreads();
  for (int off = 128; off; off >>= 1) { if (tid < off) red[tid] += red[tid + off]; __syncthreads(); }
  const float inv = rsqrtf(red[0] * (1.0f / HID) + 1e-5f);
#pragma unroll
  for (int j = 0; j < 3; ++j) {
    int i = tid + j * 256;
    float o = (vals[j] - mean) * inv * sc[i] + bs[i];
    Xf[(size_t)row * HID + i] = o;
    Xh[(size_t)row * HID + i] = f2bf(o);
  }
}

// ---------------- gather CLS rows (token 0 of each batch) ----------------
__global__ __launch_bounds__(256) void lf_gather_cls(const u16* __restrict__ Xh,
                                                     u16* __restrict__ cls) {
  int i = blockIdx.x * 256 + threadIdx.x;
  if (i < BB * HID) {
    int b = i / HID, d = i % HID;
    cls[i] = Xh[(size_t)b * SEQ * HID + d];
  }
}

// ---------------- banded sliding-window attention (flash-style) ----------------
#define KT_STRIDE 776
#define VS_STRIDE 72
#define P_STRIDE  40
#define ATTN_LDS_U16 (64 * KT_STRIDE + 3 * WW * VS_STRIDE + 8 * 16 * P_STRIDE + 128)

__global__ __launch_bounds__(256) void lf_band_attn(
    const u16* __restrict__ q, const u16* __restrict__ k, const u16* __restrict__ v,
    const u16* __restrict__ kg, const u16* __restrict__ vg,
    const int* __restrict__ am, u16* __restrict__ outb) {
  extern __shared__ u16 sm[];
  u16* Kt  = sm;                                   // K^T: [64][776]
  u16* Vs  = Kt + 64 * KT_STRIDE;                  // V:   [768][72]
  u16* Pst = Vs + 3 * WW * VS_STRIDE;              // P staging: 8 waves x [16][40]
  u16* kg0 = Pst + 8 * 16 * P_STRIDE;              // [64]
  u16* vg0 = kg0 + 64;                             // [64]
  const int tid = threadIdx.x, wv = tid >> 5, lane = tid & 31;
  int bidx = blockIdx.x;
  const int c = bidx % NC; bidx /= NC;
  const int h = bidx % HEADS; const int b = bidx / HEADS;
  const size_t baseTok = (size_t)b * SEQ;

  // stage K^T (dim-major) and V (key-major) for the 3W key band, 16B packets
  for (int kk = tid; kk < 3 * WW; kk += 256) {
    const int keypos = (c - 1) * WW + kk;
    u16* vdst = Vs + kk * VS_STRIDE;
    if (keypos >= 0 && keypos < SEQ) {
      const u16* kr = k + ((baseTok + keypos) * HID + h * DH);
      const u16* vr = v + ((baseTok + keypos) * HID + h * DH);
#pragma unroll
      for (int d8 = 0; d8 < 8; ++d8) cp16(vdst + d8 * 8, vr + d8 * 8);
#pragma unroll
      for (int d8 = 0; d8 < 8; ++d8) {
        v8u kv = *(const v8u*)(kr + d8 * 8);
#pragma unroll
        for (int e = 0; e < 8; ++e) Kt[(d8 * 8 + e) * KT_STRIDE + kk] = kv[e];
      }
    } else {
      v8u z = (v8u){};
#pragma unroll
      for (int d8 = 0; d8 < 8; ++d8) *(v8u*)(vdst + d8 * 8) = z;
#pragma unroll
      for (int d = 0; d < DH; ++d) Kt[d * KT_STRIDE + kk] = 0;
    }
  }
  if (tid < DH) {
    kg0[tid] = kg[baseTok * HID + h * DH + tid];
    vg0[tid] = vg[baseTok * HID + h * DH + tid];
  }
  lf_wait_async();
  __syncthreads();

  const float scale = 0.125f;  // 1/sqrt(64)
  u16* Pw = Pst + wv * 16 * P_STRIDE;

  for (int sIdx = 0; sIdx < 2; ++sIdx) {           // two 16-row query stripes/wave
    const int p0 = wv * 32 + sIdx * 16;
    v16bf qf[2];
#pragma unroll
    for (int ks = 0; ks < 2; ++ks) {               // Q A-fragments (16x32 each)
      size_t off = (baseTok + (size_t)c * WW + p0 + (lane & 15)) * HID +
                   h * DH + ks * 32 + ((lane >> 4) << 3);
      qf[ks] = mkfrag(*(const v8u*)(q + off), *(const v8u*)(q + off + 16));
    }
    v8f Oacc[4];
#pragma unroll
    for (int u = 0; u < 4; ++u) Oacc[u] = (v8f){};
    float mrow[8], lrow[8];
#pragma unroll
    for (int r = 0; r < 8; ++r) { mrow[r] = -1e30f; lrow[r] = 0.f; }

    for (int kp = 0; kp < 24; ++kp) {              // 24 x 32-key pairs = 768 keys
      v8f s0 = (v8f){}, s1 = (v8f){};
#pragma unroll
      for (int ks = 0; ks < 2; ++ks) {             // scores: Q . K^T
        const u16* bp = Kt + (ks * 32) * KT_STRIDE + kp * 32 + (lane & 15);
        v16bf f0, f1;
#pragma unroll
        for (int e = 0; e < 16; ++e) {
          const int kd = ((lane >> 4) << 3) + (e & 7) + ((e >> 3) << 4);
          f0[e] = __builtin_bit_cast(__bf16, bp[kd * KT_STRIDE]);
          f1[e] = __builtin_bit_cast(__bf16, bp[kd * KT_STRIDE + 16]);
        }
        s0 = wmma_bf16(qf[ks], f0, s0);
        s1 = wmma_bf16(qf[ks], f1, s1);
      }
      const int kk0 = kp * 32 + (lane & 15), kk1 = kk0 + 16;
      const int pos0 = (c - 1) * WW + kk0, pos1 = (c - 1) * WW + kk1;
      bool ok0 = (pos0 >= 0) && (pos0 < SEQ) && (pos0 != 0);
      if (ok0) ok0 = am[b * SEQ + pos0] > 0;
      bool ok1 = (pos1 >= 0) && (pos1 < SEQ) && (pos1 != 0);
      if (ok1) ok1 = am[b * SEQ + pos1] > 0;

#pragma unroll
      for (int r = 0; r < 8; ++r) {
        const int p = p0 + r + ((lane >> 4) << 3);
        float sv0 = (ok0 && kk0 >= p && kk0 <= p + 2 * WW) ? s0[r] * scale : NEGBIG;
        float sv1 = (ok1 && kk1 >= p && kk1 <= p + 2 * WW) ? s1[r] * scale : NEGBIG;
        float tmax = fmaxf(sv0, sv1);
        tmax = fmaxf(tmax, __shfl_xor(tmax, 1));
        tmax = fmaxf(tmax, __shfl_xor(tmax, 2));
        tmax = fmaxf(tmax, __shfl_xor(tmax, 4));
        tmax = fmaxf(tmax, __shfl_xor(tmax, 8));
        const float nm = fmaxf(mrow[r], tmax);
        const float corr = __expf(mrow[r] - nm);
        const float e0 = __expf(sv0 - nm), e1 = __expf(sv1 - nm);
        float ps = e0 + e1;
        ps += __shfl_xor(ps, 1); ps += __shfl_xor(ps, 2);
        ps += __shfl_xor(ps, 4); ps += __shfl_xor(ps, 8);
        lrow[r] = lrow[r] * corr + ps;
        mrow[r] = nm;
#pragma unroll
        for (int u = 0; u < 4; ++u) Oacc[u][r] *= corr;
        const int prw = (r + ((lane >> 4) << 3)) * P_STRIDE + (lane & 15);
        Pw[prw]      = f2bf(e0);                   // stage P (C-layout -> LDS)
        Pw[prw + 16] = f2bf(e1);
      }
      // PV: reload P as A-fragment (LDS in-order within wave), V as B-fragment
      const u16* pr = Pw + (lane & 15) * P_STRIDE + ((lane >> 4) << 3);
      v16bf pf = mkfrag(*(const v8u*)pr, *(const v8u*)(pr + 16));
#pragma unroll
      for (int u = 0; u < 4; ++u) {
        const u16* vp = Vs + (kp * 32) * VS_STRIDE + u * 16 + (lane & 15);
        v16bf vf;
#pragma unroll
        for (int e = 0; e < 16; ++e)
          vf[e] = __builtin_bit_cast(__bf16, vp[(((lane >> 4) << 3) + (e & 7) + ((e >> 3) << 4)) * VS_STRIDE]);
        Oacc[u] = wmma_bf16(pf, vf, Oacc[u]);
      }
    }

    // global (CLS) key column: score = q . kg[0]
    float part = 0.f;
#pragma unroll
    for (int ks = 0; ks < 2; ++ks)
#pragma unroll
      for (int e = 0; e < 16; ++e) {
        const int d = ks * 32 + ((lane >> 4) << 3) + (e & 7) + ((e >> 3) << 4);
        part += bf2f(__builtin_bit_cast(u16, qf[ks][e])) * bf2f(kg0[d]);
      }
    const float gAll = part + __shfl_xor(part, 16);  // lane holds g of row (lane&15)
#pragma unroll
    for (int r = 0; r < 8; ++r) {
      const int src = r + ((lane >> 4) << 3);
      const float gs = __shfl(gAll, src) * scale;
      const float nm = fmaxf(mrow[r], gs);
      const float corr = __expf(mrow[r] - nm);
      const float pg = __expf(gs - nm);
      lrow[r] = lrow[r] * corr + pg;
      mrow[r] = nm;
#pragma unroll
      for (int u = 0; u < 4; ++u)
        Oacc[u][r] = Oacc[u][r] * corr + pg * bf2f(vg0[u * 16 + (lane & 15)]);
    }

    // normalize + store, layout [b, s, h*64+d]
#pragma unroll
    for (int u = 0; u < 4; ++u)
#pragma unroll
      for (int r = 0; r < 8; ++r) {
        const int row = p0 + r + ((lane >> 4) << 3);
        const size_t tok = baseTok + (size_t)c * WW + row;
        const int col = h * DH + u * 16 + (lane & 15);
        outb[tok * HID + col] = f2bf(Oacc[u][r] / lrow[r]);
      }
  }
}

// ---------------- global (CLS) query full attention per (b,h) ----------------
__global__ __launch_bounds__(256) void lf_global_attn(
    const u16* __restrict__ qgc, const u16* __restrict__ kg, const u16* __restrict__ vg,
    const int* __restrict__ am, u16* __restrict__ outb) {
  __shared__ float sc[SEQ];
  __shared__ float red[256];
  __shared__ float qs[DH];
  const int tid = threadIdx.x;
  const int h = blockIdx.x % HEADS, b = blockIdx.x / HEADS;
  if (tid < DH) qs[tid] = bf2f(qgc[b * HID + h * DH + tid]);
  __syncthreads();
  float lmax = -1e30f;
  for (int j = 0; j < 16; ++j) {
    const int s = tid + j * 256;
    const v8u* kr8 = (const v8u*)(kg + ((size_t)b * SEQ + s) * HID + h * DH);
    float acc = 0.f;
#pragma unroll
    for (int d8 = 0; d8 < 8; ++d8) {
      v8u kv = kr8[d8];
#pragma unroll
      for (int e = 0; e < 8; ++e) acc += qs[d8 * 8 + e] * bf2f(kv[e]);
    }
    acc *= 0.125f;
    if (am[b * SEQ + s] <= 0) acc = NEGBIG;
    sc[s] = acc; lmax = fmaxf(lmax, acc);
  }
  red[tid] = lmax; __syncthreads();
  for (int off = 128; off; off >>= 1) { if (tid < off) red[tid] = fmaxf(red[tid], red[tid + off]); __syncthreads(); }
  const float gmax = red[0]; __syncthreads();
  float lsum = 0.f;
  for (int j = 0; j < 16; ++j) {
    const int s = tid + j * 256;
    const float p = __expf(sc[s] - gmax);
    sc[s] = p; lsum += p;
  }
  red[tid] = lsum; __syncthreads();
  for (int off = 128; off; off >>= 1) { if (tid < off) red[tid] += red[tid + off]; __syncthreads(); }
  const float inv = 1.0f / red[0]; __syncthreads();
  const int d = tid & 63, ch = tid >> 6;           // 4 key chunks x 64 dims
  float acc = 0.f;
  for (int s = ch * 1024; s < (ch + 1) * 1024; ++s)
    acc += sc[s] * bf2f(vg[((size_t)b * SEQ + s) * HID + h * DH + d]);
  red[tid] = acc; __syncthreads();
  if (tid < 64) {
    const float o = (red[tid] + red[tid + 64] + red[tid + 128] + red[tid + 192]) * inv;
    outb[(size_t)b * SEQ * HID + h * DH + tid] = f2bf(o);
  }
}

// ---------------- host-side orchestration ----------------
static inline void launch_gemm(const u16* A, const u16* Bw, const float* bias,
                               float* Cf, u16* Cb, int M, int N, int K, int ldB,
                               int mode, hipStream_t st) {
  dim3 g((N + 127) / 128, (M + 63) / 64);
  lf_gemm_bf16<<<g, dim3(256), 0, st>>>(A, Bw, bias, Cf, Cb, M, N, K, ldB, mode);
}
static inline void launch_cvt(const float* x, u16* y, long n, hipStream_t st) {
  lf_cvt_bf16<<<(unsigned)((n + 255) / 256), 256, 0, st>>>(x, y, n);
}

extern "C" void kernel_launch(void* const* d_in, const int* in_sizes, int n_in,
                              void* d_out, int out_size, void* d_ws, size_t ws_size,
                              hipStream_t stream) {
  (void)in_sizes; (void)n_in; (void)out_size; (void)ws_size;
  // inputs, in setup_inputs() order
  const int*   ids   = (const int*)  d_in[0];
  const int*   am    = (const int*)  d_in[1];
  const float* wemb  = (const float*)d_in[2];
  const float* pemb  = (const float*)d_in[3];
  const float* elns  = (const float*)d_in[4];
  const float* elnb  = (const float*)d_in[5];
  const float* Wq    = (const float*)d_in[6];  const float* bq  = (const float*)d_in[7];
  const float* Wk    = (const float*)d_in[8];  const float* bk  = (const float*)d_in[9];
  const float* Wv    = (const float*)d_in[10]; const float* bv  = (const float*)d_in[11];
  const float* Wqg   = (const float*)d_in[12]; const float* bqg = (const float*)d_in[13];
  const float* Wkg   = (const float*)d_in[14]; const float* bkg = (const float*)d_in[15];
  const float* Wvg   = (const float*)d_in[16]; const float* bvg = (const float*)d_in[17];
  const float* Wo    = (const float*)d_in[18]; const float* bo  = (const float*)d_in[19];
  const float* ln1s  = (const float*)d_in[20]; const float* ln1b = (const float*)d_in[21];
  const float* W1    = (const float*)d_in[22]; const float* b1  = (const float*)d_in[23];
  const float* W2    = (const float*)d_in[24]; const float* b2  = (const float*)d_in[25];
  const float* ln2s  = (const float*)d_in[26]; const float* ln2b = (const float*)d_in[27];
  const float* hW    = (const float*)d_in[28]; const float* hb  = (const float*)d_in[29];
  float* out = (float*)d_out;

  // workspace carve-out (256B aligned)
  char* base = (char*)d_ws; size_t off = 0;
  auto alloc = [&](size_t bytes) -> void* {
    void* r = base + off; off = (off + bytes + 255) & ~(size_t)255; return r;
  };
  const size_t PW = (size_t)LAYERS * HID * HID;      // per-projection weight elems
  const size_t FW = (size_t)LAYERS * HID * FFN;
  u16* wqh  = (u16*)alloc(PW * 2); u16* wkh  = (u16*)alloc(PW * 2);
  u16* wvh  = (u16*)alloc(PW * 2); u16* wqgh = (u16*)alloc(PW * 2);
  u16* wkgh = (u16*)alloc(PW * 2); u16* wvgh = (u16*)alloc(PW * 2);
  u16* woh  = (u16*)alloc(PW * 2);
  u16* w1h  = (u16*)alloc(FW * 2); u16* w2h  = (u16*)alloc(FW * 2);
  u16* whh  = (u16*)alloc((size_t)HID * NLABP * 2);  // padded head weights
  float* Xf  = (float*)alloc((size_t)TOK * HID * 4);
  float* yo  = (float*)alloc((size_t)TOK * HID * 4);
  u16* Xh    = (u16*)alloc((size_t)TOK * HID * 2);
  u16* qb    = (u16*)alloc((size_t)TOK * HID * 2);
  u16* kb    = (u16*)alloc((size_t)TOK * HID * 2);
  u16* vb    = (u16*)alloc((size_t)TOK * HID * 2);
  u16* kgb   = (u16*)alloc((size_t)TOK * HID * 2);
  u16* vgb   = (u16*)alloc((size_t)TOK * HID * 2);
  u16* attnb = (u16*)alloc((size_t)TOK * HID * 2);
  u16* Hb    = (u16*)alloc((size_t)TOK * FFN * 2);
  u16* qgcls = (u16*)alloc((size_t)BB * HID * 2);
  u16* clsX  = (u16*)alloc((size_t)BB * HID * 2);

  // weights fp32 -> bf16 (every call; no cached state allowed)
  launch_cvt(Wq,  wqh,  PW, stream); launch_cvt(Wk,  wkh,  PW, stream);
  launch_cvt(Wv,  wvh,  PW, stream); launch_cvt(Wqg, wqgh, PW, stream);
  launch_cvt(Wkg, wkgh, PW, stream); launch_cvt(Wvg, wvgh, PW, stream);
  launch_cvt(Wo,  woh,  PW, stream);
  launch_cvt(W1,  w1h,  FW, stream); launch_cvt(W2,  w2h,  FW, stream);
  {
    long n = (long)HID * NLABP;
    lf_cvt_pad_bf16<<<(unsigned)((n + 255) / 256), 256, 0, stream>>>(hW, whh, HID, NLAB, NLABP);
  }

  // embeddings + LN
  lf_embed_add<<<TOK, 256, 0, stream>>>(ids, wemb, pemb, yo);
  lf_resid_ln<<<TOK, 256, 0, stream>>>(nullptr, yo, elns, elnb, Xf, Xh);

  const size_t attn_lds = (size_t)ATTN_LDS_U16 * 2;  // ~220 KB dynamic LDS (WGP has 320 KB)
  for (int l = 0; l < LAYERS; ++l) {
    const size_t wO = (size_t)l * HID * HID;
    const size_t fO = (size_t)l * HID * FFN;
    launch_gemm(Xh, wqh  + wO, bq  + l * HID, nullptr, qb,  TOK, HID, HID, HID, 1, stream);
    launch_gemm(Xh, wkh  + wO, bk  + l * HID, nullptr, kb,  TOK, HID, HID, HID, 1, stream);
    launch_gemm(Xh, wvh  + wO, bv  + l * HID, nullptr, vb,  TOK, HID, HID, HID, 1, stream);
    launch_gemm(Xh, wkgh + wO, bkg + l * HID, nullptr, kgb, TOK, HID, HID, HID, 1, stream);
    launch_gemm(Xh, wvgh + wO, bvg + l * HID, nullptr, vgb, TOK, HID, HID, HID, 1, stream);
    // qg is only consumed at the CLS token -> compute just 2 rows
    lf_gather_cls<<<(BB * HID + 255) / 256, 256, 0, stream>>>(Xh, clsX);
    launch_gemm(clsX, wqgh + wO, bqg + l * HID, nullptr, qgcls, BB, HID, HID, HID, 1, stream);

    lf_band_attn<<<BB * HEADS * NC, 256, attn_lds, stream>>>(qb, kb, vb, kgb, vgb, am, attnb);
    lf_global_attn<<<BB * HEADS, 256, 0, stream>>>(qgcls, kgb, vgb, am, attnb);

    launch_gemm(attnb, woh + wO, bo + l * HID, yo, nullptr, TOK, HID, HID, HID, 0, stream);
    lf_resid_ln<<<TOK, 256, 0, stream>>>(Xf, yo, ln1s + l * HID, ln1b + l * HID, Xf, Xh);

    launch_gemm(Xh, w1h + fO, b1 + l * FFN, nullptr, Hb, TOK, FFN, HID, FFN, 2, stream);  // GELU
    launch_gemm(Hb, w2h + fO, b2 + l * HID, yo, nullptr, TOK, HID, FFN, HID, 0, stream);
    lf_resid_ln<<<TOK, 256, 0, stream>>>(Xf, yo, ln2s + l * HID, ln2b + l * HID, Xf, Xh);
  }

  // classification head on CLS tokens (B is zero-padded to ld = 8960)
  lf_gather_cls<<<(BB * HID + 255) / 256, 256, 0, stream>>>(Xh, clsX);
  launch_gemm(clsX, whh, hb, out, nullptr, BB, NLAB, HID, NLABP, 0, stream);
}
